// MoEGate_27041114096167
// MI455X (gfx1250) — compile-verified, matrix operand order checked by reading, not run on hardware
//
#include <hip/hip_runtime.h>
#include <math.h>

// ---------------------------------------------------------------------------
// MoE gate for MI455X (gfx1250): logits = x @ W^T, softmax, top-2.
//   x: [32768, 4096] f32   W: [64, 4096] f32
// bf16 hi/lo split GEMM on v_wmma_f32_16x16x32_bf16 (3 WMMAs per tile pair
// ~ fp32 accuracy). M=32 tokens per wave (2 M-tiles) so each B fragment
// read is amortized over twice the A-work (halves WGP$/L2 B bandwidth).
// Weights are pre-swizzled once into exact B-fragment layout in d_ws.
// ---------------------------------------------------------------------------

typedef __attribute__((ext_vector_type(16))) __bf16 v16bf;
typedef __attribute__((ext_vector_type(8)))  float  v8f;

constexpr int T      = 4 * 8192;   // 32768 tokens
constexpr int H      = 4096;       // hidden
constexpr int NE     = 64;         // experts
constexpr int KTILES = H / 32;     // 128 k-tiles of 32
// ws layout: fragment (t, e) occupies 1024 bf16 (2 KB):
//   hi half: lane*16 + p   lo half: 512 + lane*16 + p
// lane n   (n<16) holds expert col n,  K in {0..7, 16..23} of the 32-chunk
// lane n+16       holds expert col n,  K in {8..15, 24..31}

__global__ void pack_weight_bf16x2(const float* __restrict__ W,
                                   __bf16* __restrict__ WB) {
    int u = blockIdx.x * blockDim.x + threadIdx.x;   // [0, NE*H/2)
    int efull = u >> 11;          // H/2 = 2048 k-pairs per expert
    int j     = u & 2047;
    int k0    = j << 1;           // even k
    int t     = k0 >> 5;
    int kin   = k0 & 31;
    int et    = efull >> 4;
    int ncol  = efull & 15;

    int lane, p;
    if (kin < 8)       { lane = ncol;      p = kin;      }
    else if (kin < 16) { lane = ncol + 16; p = kin - 8;  }
    else if (kin < 24) { lane = ncol;      p = kin - 8;  }
    else               { lane = ncol + 16; p = kin - 16; }

    float f0 = W[(size_t)efull * H + k0];
    float f1 = W[(size_t)efull * H + k0 + 1];
    __bf16 h0 = (__bf16)f0, h1 = (__bf16)f1;
    __bf16 l0 = (__bf16)(f0 - (float)h0);
    __bf16 l1 = (__bf16)(f1 - (float)h1);

    int base = ((t << 2) + et) << 10;   // (t*4+e)*1024 bf16
    int off  = base + lane * 16 + p;
    WB[off]           = h0;
    WB[off + 1]       = h1;
    WB[off + 512]     = l0;
    WB[off + 512 + 1] = l1;
}

__device__ __forceinline__ void make_frags(const float4& f0, const float4& f1,
                                           const float4& f2, const float4& f3,
                                           v16bf& a_hi, v16bf& a_lo) {
    const float fa[16] = {f0.x,f0.y,f0.z,f0.w, f1.x,f1.y,f1.z,f1.w,
                          f2.x,f2.y,f2.z,f2.w, f3.x,f3.y,f3.z,f3.w};
    #pragma unroll
    for (int j = 0; j < 16; ++j) {
        __bf16 h = (__bf16)fa[j];
        a_hi[j]  = h;
        a_lo[j]  = (__bf16)(fa[j] - (float)h);
    }
}

__global__ __launch_bounds__(256) void moe_gate_kernel(
        const float* __restrict__ X, const __bf16* __restrict__ WB,
        int* __restrict__ idx_out, float* __restrict__ w_out) {
    const int wave   = (int)((blockIdx.x * blockDim.x + threadIdx.x) >> 5);
    const int lane   = threadIdx.x & 31;
    const int half   = lane >> 4;
    const int n      = lane & 15;
    const int token0 = wave * 32;            // 2 M-tiles of 16 tokens

    // Per-lane contiguous float4 loads of each token row.
    const float4* row0 = (const float4*)(X + (size_t)(token0 + n) * H);
    const float4* row1 = (const float4*)(X + (size_t)(token0 + 16 + n) * H);
    const v16bf*  bp   = (const v16bf*)WB;   // 32B fragments

    v8f acc[2][4];
    #pragma unroll
    for (int m = 0; m < 2; ++m)
        #pragma unroll
        for (int e = 0; e < 4; ++e)
            acc[m][e] = (v8f){0.f,0.f,0.f,0.f,0.f,0.f,0.f,0.f};

    #pragma unroll 1
    for (int t = 0; t < KTILES; ++t) {
        const int q = t * 8 + half * 2;      // float4 index of segment 0
        float4 a0 = row0[q];
        float4 a1 = row0[q + 1];
        float4 a2 = row0[q + 4];             // segment 1 (+16 K)
        float4 a3 = row0[q + 5];
        float4 b0 = row1[q];
        float4 b1 = row1[q + 1];
        float4 b2 = row1[q + 4];
        float4 b3 = row1[q + 5];
        __builtin_prefetch(row0 + q + 8, 0, 0);
        __builtin_prefetch(row1 + q + 8, 0, 0);

        v16bf a_hi0, a_lo0, a_hi1, a_lo1;
        make_frags(a0, a1, a2, a3, a_hi0, a_lo0);
        make_frags(b0, b1, b2, b3, a_hi1, a_lo1);

        #pragma unroll
        for (int e = 0; e < 4; ++e) {
            const int fb = ((t * 4 + e) << 6) + lane;  // v16bf units
            v16bf b_hi = bp[fb];
            v16bf b_lo = bp[fb + 32];
            acc[0][e] = __builtin_amdgcn_wmma_f32_16x16x32_bf16(
                false, a_hi0, false, b_hi, (short)0, acc[0][e], false, false);
            acc[0][e] = __builtin_amdgcn_wmma_f32_16x16x32_bf16(
                false, a_lo0, false, b_hi, (short)0, acc[0][e], false, false);
            acc[0][e] = __builtin_amdgcn_wmma_f32_16x16x32_bf16(
                false, a_hi0, false, b_lo, (short)0, acc[0][e], false, false);
            acc[1][e] = __builtin_amdgcn_wmma_f32_16x16x32_bf16(
                false, a_hi1, false, b_hi, (short)0, acc[1][e], false, false);
            acc[1][e] = __builtin_amdgcn_wmma_f32_16x16x32_bf16(
                false, a_lo1, false, b_hi, (short)0, acc[1][e], false, false);
            acc[1][e] = __builtin_amdgcn_wmma_f32_16x16x32_bf16(
                false, a_hi1, false, b_lo, (short)0, acc[1][e], false, false);
        }
    }

    // Epilogue. C layout: lane (half h, col n) holds logit(tile_base + 8h + i,
    // expert e*16+n) in acc[m][e][i]. Reduce over the 16 lanes of each half.
    #pragma unroll
    for (int m = 0; m < 2; ++m) {
        #pragma unroll
        for (int i = 0; i < 8; ++i) {
            float av[4];
            #pragma unroll
            for (int e = 0; e < 4; ++e) av[e] = acc[m][e][i];

            // max over 64 logits
            float mx = fmaxf(fmaxf(av[0], av[1]), fmaxf(av[2], av[3]));
            #pragma unroll
            for (int mask = 1; mask <= 8; mask <<= 1)
                mx = fmaxf(mx, __shfl_xor(mx, mask));

            // sum of exp
            float s = expf(av[0]-mx) + expf(av[1]-mx)
                    + expf(av[2]-mx) + expf(av[3]-mx);
            #pragma unroll
            for (int mask = 1; mask <= 8; mask <<= 1)
                s += __shfl_xor(s, mask);

            // local top-2 (lower index wins ties, matching lax.top_k)
            float v1 = -INFINITY, v2 = -INFINITY;
            int   j1 = 1 << 30,   j2 = 1 << 30;
            #pragma unroll
            for (int e = 0; e < 4; ++e) {
                float v = av[e]; int id = e * 16 + n;
                if ((v > v1) || (v == v1 && id < j1)) {
                    v2 = v1; j2 = j1; v1 = v; j1 = id;
                } else if ((v > v2) || (v == v2 && id < j2)) {
                    v2 = v; j2 = id;
                }
            }
            // butterfly merge of sorted pairs across 16 lanes
            #pragma unroll
            for (int mask = 1; mask <= 8; mask <<= 1) {
                float o1 = __shfl_xor(v1, mask); int oj1 = __shfl_xor(j1, mask);
                float o2 = __shfl_xor(v2, mask); int oj2 = __shfl_xor(j2, mask);
                if ((o1 > v1) || (o1 == v1 && oj1 < j1)) {
                    if ((v1 > o2) || (v1 == o2 && j1 < oj2)) { v2 = v1; j2 = j1; }
                    else                                     { v2 = o2; j2 = oj2; }
                    v1 = o1; j1 = oj1;
                } else if ((o1 > v2) || (o1 == v2 && oj1 < j2)) {
                    v2 = o1; j2 = oj1;
                }
            }

            if (n == 0) {
                const int token = token0 + m * 16 + half * 8 + i;
                idx_out[2 * token]     = j1;
                idx_out[2 * token + 1] = j2;
                const float inv = 1.0f / s;
                w_out[2 * token]       = expf(v1 - mx) * inv;
                w_out[2 * token + 1]   = expf(v2 - mx) * inv;
            }
        }
    }
}

extern "C" void kernel_launch(void* const* d_in, const int* in_sizes, int n_in,
                              void* d_out, int out_size, void* d_ws, size_t ws_size,
                              hipStream_t stream) {
    const float* X = (const float*)d_in[0];   // hidden_states [4,8192,4096] f32
    const float* W = (const float*)d_in[1];   // weight        [64,4096]     f32
    __bf16* WB = (__bf16*)d_ws;               // 1 MB packed hi/lo B fragments

    // One-time (per call) weight repack: 64*4096/2 threads.
    pack_weight_bf16x2<<<(NE * H / 2) / 256, 256, 0, stream>>>(W, WB);

    int*   idx_out = (int*)d_out;             // [T, 2] int32
    float* w_out   = (float*)d_out + (size_t)T * 2;  // [T, 2] f32

    // 1024 waves, one 32-token tile each: 128 blocks x 8 waves.
    moe_gate_kernel<<<(T / 32) / 8, 256, 0, stream>>>(X, WB, idx_out, w_out);
}
